// GCN_67894843015748
// MI455X (gfx1250) — matrix operand order repaired and checked by reading, not compiled, not run on hardware
//
#include <hip/hip_runtime.h>

// ---------------------------------------------------------------------------
// GCN (2-layer) on gfx1250:  h = relu(S x W1 + b1); out = S h W2 + b2
// S = sym-normalized adjacency with self loops.
// GEMMs: v_wmma_f32_16x16x32_bf16, 64 rows/block, B-fragments register-resident
// across 4 row tiles. Edge pass: wave32-per-edge float4 gather +
// global_atomic_add_f32 scatter (L2-resident: t/acc buffers << 192MB L2).
// ---------------------------------------------------------------------------

typedef __attribute__((ext_vector_type(16))) __bf16 v16bf;
typedef __attribute__((ext_vector_type(8)))  float  v8f;

union Frag16 { v16bf v; uint4 q[2]; };

__device__ __forceinline__ unsigned short f2bf(float f) {
    unsigned u = __float_as_uint(f);
    u += 0x7FFFu + ((u >> 16) & 1u);          // round-to-nearest-even
    return (unsigned short)(u >> 16);
}

// ---------------- degree / normalization -----------------------------------
__global__ void gcn_fill_one(float* __restrict__ deg, int n) {
    int i = blockIdx.x * blockDim.x + threadIdx.x;
    if (i < n) deg[i] = 1.0f;                 // self-loop contributes 1
}

__global__ void gcn_deg_accum(float* __restrict__ deg, const int* __restrict__ ei,
                              int E) {
    int e = blockIdx.x * blockDim.x + threadIdx.x;
    if (e < E) atomicAdd(&deg[ei[E + e]], 1.0f);   // segment_sum over col
}

__global__ void gcn_rsqrt(float* __restrict__ deg, int n) {
    int i = blockIdx.x * blockDim.x + threadIdx.x;
    if (i < n) deg[i] = rsqrtf(deg[i]);       // deg >= 1 always
}

// ---------------- f32 -> bf16 conversions ----------------------------------
__global__ void gcn_cvt_bf16(const float* __restrict__ x,
                             unsigned short* __restrict__ y, int n) {
    int i = blockIdx.x * blockDim.x + threadIdx.x;
    if (i < n) y[i] = f2bf(x[i]);
}

// W is [K=128][Ncols] f32 row-major; emit Wt [Ncols][128] bf16 (K contiguous)
__global__ void gcn_cvt_w_t(const float* __restrict__ W,
                            unsigned short* __restrict__ Wt, int ncols, int total) {
    int i = blockIdx.x * blockDim.x + threadIdx.x;   // i = n*128 + k
    if (i < total) {
        int n = i >> 7, k = i & 127;
        Wt[i] = f2bf(W[k * ncols + n]);
    }
}

// ---------------- WMMA GEMM: C[M x NCOLS] = A[M x 128] * Wt^T --------------
// 64 rows per block; blockDim.x = (NCOLS/16)*32, one wave per 16-col tile.
// B k-fragments loaded once per wave, reused across 4 row tiles (16 WMMA/wave).
template <int NCOLS>
__global__ void __launch_bounds__(256)
gcn_gemm_bf16(const unsigned short* __restrict__ A,   // M x 128 bf16 row-major
              const unsigned short* __restrict__ Wt,  // NCOLS x 128 bf16
              float* __restrict__ C, int M) {
    __shared__ __align__(16) unsigned char lds[64 * 128 * 2];  // 16KB A stripe

    const int lane    = threadIdx.x & 31;
    const int wave    = threadIdx.x >> 5;
    const int rowBase = blockIdx.x * 64;
    const int ntiles  = M >> 4;                   // M % 16 == 0 (N = 50000)

    // cooperative stage of up to 64x128 bf16 rows (1024 x uint4)
    uint4* dstl = (uint4*)lds;
    for (int t = threadIdx.x; t < 1024; t += blockDim.x) {
        int srcRow = rowBase + (t >> 4);          // tile row 0..63
        if (srcRow >= M) srcRow = M - 1;          // clamp (last partial block)
        dstl[t] = ((const uint4*)(A + srcRow * 128))[t & 15];
    }
    __syncthreads();

    const int m  = lane & 15;                     // A row within tile / B col
    const int hs = lane >> 4;                     // half-select (K grouping)
    const int colBase = wave * 16;

    // B k order used here (contiguous, same as A): lanes 0-15 carry K 0..15, lanes 16-31 K 16..31
    // (contiguous in Wt's [N][K] layout) -- load all 4 k-fragments once.
    const unsigned char* wrow =
        (const unsigned char*)(Wt + (colBase + m) * 128);
    Frag16 b[4];
#pragma unroll
    for (int k = 0; k < 4; ++k) {
        const unsigned char* bbase = wrow + hs * 32 + k * 64;
        b[k].q[0] = *(const uint4*)(bbase);
        b[k].q[1] = *(const uint4*)(bbase + 16);
    }

#pragma unroll
    for (int rt = 0; rt < 4; ++rt) {
        const int tile = blockIdx.x * 4 + rt;
        if (tile >= ntiles) break;                // uniform scalar branch
        v8f acc = {};
#pragma unroll
        for (int k = 0; k < 4; ++k) {             // K = 4 * 32
            Frag16 a;
            // A k order used here (contiguous, same as B): lanes 0-15 K 0..15, lanes 16-31 K 16..31
            const unsigned char* abase =
                lds + (rt * 16 + m) * 256 + k * 64 + hs * 32;
            a.q[0] = *(const uint4*)(abase);
            a.q[1] = *(const uint4*)(abase + 16);
            acc = __builtin_amdgcn_wmma_f32_16x16x32_bf16(
                false, a.v, false, b[k].v, (short)0, acc, false, false);
        }
        // D layout: VGPR v -> row (hs*8 + v), col = lane&15; int32 addressing
        float* crow = C + (tile * 16 + hs * 8) * NCOLS + colBase + m;
#pragma unroll
        for (int v = 0; v < 8; ++v) crow[v * NCOLS] = acc[v];
    }
}

// ---------------- edge scatter: wave32 per edge ----------------------------
__global__ void gcn_scatter128(const int* __restrict__ ei, int E,
                               const float* __restrict__ dis,
                               const float* __restrict__ T,    // N x 128
                               float* __restrict__ ACC) {      // N x 128
    const int lane = threadIdx.x & 31;
    const int e = blockIdx.x * (blockDim.x >> 5) + (threadIdx.x >> 5);
    if (e >= E) return;
    const int r = ei[e], c = ei[E + e];
    const float nrm = dis[r] * dis[c];
    float4 v = ((const float4*)T)[r * 32 + lane];
    float* dst = ACC + c * 128 + lane * 4;
    atomicAdd(dst + 0, v.x * nrm);
    atomicAdd(dst + 1, v.y * nrm);
    atomicAdd(dst + 2, v.z * nrm);
    atomicAdd(dst + 3, v.w * nrm);
}

__global__ void gcn_scatter64(const int* __restrict__ ei, int E,
                              const float* __restrict__ dis,
                              const float* __restrict__ T,     // N x 64
                              float* __restrict__ ACC) {       // N x 64
    const int lane = threadIdx.x & 31;
    const int e = blockIdx.x * (blockDim.x >> 5) + (threadIdx.x >> 5);
    if (e >= E) return;
    const int r = ei[e], c = ei[E + e];
    const float nrm = dis[r] * dis[c];
    float2 v = ((const float2*)T)[r * 32 + lane];
    float* dst = ACC + c * 64 + lane * 2;
    atomicAdd(dst + 0, v.x * nrm);
    atomicAdd(dst + 1, v.y * nrm);
}

// ---------------- fused epilogues ------------------------------------------
// h = relu(acc + t*dis^2 (self loop) + b1)  ->  bf16
__global__ void gcn_bias_relu_bf16(const float* __restrict__ acc,
                                   const float* __restrict__ t,
                                   const float* __restrict__ dis,
                                   const float* __restrict__ b,
                                   unsigned short* __restrict__ hb, int total) {
    int idx = blockIdx.x * blockDim.x + threadIdx.x;
    if (idx >= total) return;
    int i = idx >> 7, f = idx & 127;
    float d = dis[i];
    float v = acc[idx] + t[idx] * d * d + b[f];
    hb[idx] = f2bf(fmaxf(v, 0.0f));
}

// out += t*dis^2 (self loop) + b2
__global__ void gcn_bias_out(float* __restrict__ out,
                             const float* __restrict__ t,
                             const float* __restrict__ dis,
                             const float* __restrict__ b, int total) {
    int idx = blockIdx.x * blockDim.x + threadIdx.x;
    if (idx >= total) return;
    int i = idx >> 6, f = idx & 63;
    float d = dis[i];
    out[idx] += t[idx] * d * d + b[f];
}

// ---------------------------------------------------------------------------
extern "C" void kernel_launch(void* const* d_in, const int* in_sizes, int n_in,
                              void* d_out, int out_size, void* d_ws, size_t ws_size,
                              hipStream_t stream) {
    const float* x  = (const float*)d_in[0];
    const int*   ei = (const int*)d_in[1];
    const float* W1 = (const float*)d_in[2];
    const float* b1 = (const float*)d_in[3];
    const float* W2 = (const float*)d_in[4];
    const float* b2 = (const float*)d_in[5];

    const int Nn = in_sizes[0] / 128;   // 50000
    const int E  = in_sizes[1] / 2;     // 800000
    const int IN = 128, HID = 128, OUT = 64;

    // workspace carve-up (256B aligned slices)
    char* ws = (char*)d_ws;
    size_t off = 0;
    auto carve = [&](size_t bytes) {
        char* p = ws + off;
        off += (bytes + 255) & ~(size_t)255;
        return p;
    };
    float*          dis  = (float*)         carve((size_t)Nn * 4);            // deg -> dis
    unsigned short* xb   = (unsigned short*)carve((size_t)Nn * IN  * 2);
    unsigned short* hb   = (unsigned short*)carve((size_t)Nn * HID * 2);
    unsigned short* w1t  = (unsigned short*)carve((size_t)HID * IN * 2);
    unsigned short* w2t  = (unsigned short*)carve((size_t)OUT * HID * 2);
    float*          t1   = (float*)         carve((size_t)Nn * HID * 4);
    float*          acc1 = (float*)         carve((size_t)Nn * HID * 4);
    float*          t2   = (float*)         carve((size_t)Nn * OUT * 4);
    (void)ws_size; (void)n_in;

    float* out = (float*)d_out;

    // --- normalization: deg -> dis = rsqrt(deg) ---
    gcn_fill_one<<<(Nn + 255) / 256, 256, 0, stream>>>(dis, Nn);
    gcn_deg_accum<<<(E + 255) / 256, 256, 0, stream>>>(dis, ei, E);
    gcn_rsqrt<<<(Nn + 255) / 256, 256, 0, stream>>>(dis, Nn);

    // --- bf16 operand prep ---
    gcn_cvt_bf16<<<(Nn * IN + 255) / 256, 256, 0, stream>>>(x, xb, Nn * IN);
    gcn_cvt_w_t<<<(HID * IN + 255) / 256, 256, 0, stream>>>(W1, w1t, HID, HID * IN);
    gcn_cvt_w_t<<<(OUT * HID + 255) / 256, 256, 0, stream>>>(W2, w2t, OUT, OUT * HID);

    // --- zero accumulators (capture-safe memset nodes) ---
    hipMemsetAsync(acc1, 0, (size_t)Nn * HID * 4, stream);
    hipMemsetAsync(out,  0, (size_t)out_size * 4, stream);

    const int nblk64 = (Nn + 63) / 64;  // 64 rows per block

    // --- layer 1: t1 = x @ W1 (WMMA bf16), scatter, relu+bias -> hb ---
    gcn_gemm_bf16<128><<<nblk64, (HID / 16) * 32, 0, stream>>>(xb, w1t, t1, Nn);
    gcn_scatter128<<<(E + 7) / 8, 256, 0, stream>>>(ei, E, dis, t1, acc1);
    gcn_bias_relu_bf16<<<(Nn * HID + 255) / 256, 256, 0, stream>>>(
        acc1, t1, dis, b1, hb, Nn * HID);

    // --- layer 2: t2 = h @ W2 (WMMA bf16), scatter into out, epilogue ---
    gcn_gemm_bf16<64><<<nblk64, (OUT / 16) * 32, 0, stream>>>(hb, w2t, t2, Nn);
    gcn_scatter64<<<(E + 7) / 8, 256, 0, stream>>>(ei, E, dis, t2, out);
    gcn_bias_out<<<(Nn * OUT + 255) / 256, 256, 0, stream>>>(
        out, t2, dis, b2, Nn * OUT);
}
